// Attn_25099788878393
// MI455X (gfx1250) — matrix operand order repaired and checked by reading, not compiled, not run on hardware
//
#include <hip/hip_runtime.h>
#include <hip/hip_bf16.h>
#include <math.h>
#include <stdint.h>

typedef __attribute__((ext_vector_type(16))) _Float16 v16h;
typedef __attribute__((ext_vector_type(8)))  _Float16 v8h;
typedef __attribute__((ext_vector_type(4)))  _Float16 v4h;
typedef __attribute__((ext_vector_type(8)))  float    v8f;

#define B_   2
#define L_   2048
#define D_   2048
#define H_   16
#define HD_  128
#define TOK  (B_*L_)     /* 4096 */
#define N3   (3*D_)      /* 6144 */

#define WMMA_F16(a,b,c) \
  __builtin_amdgcn_wmma_f32_16x16x32_f16(false,(a),false,(b),(short)0,(c),false,false)

__device__ __forceinline__ v16h hcat(v8h lo, v8h hi) {
  v16h r;
#pragma unroll
  for (int i = 0; i < 8; ++i) { r[i] = lo[i]; r[i + 8] = hi[i]; }
  return r;
}

// Async DMA: copy 64 bytes (4 x b128; the shared IOFFSET is applied to both the
// global and the LDS address) from global memory to LDS, tracked by ASYNCcnt.
__device__ __forceinline__ void async_copy64(uint32_t lds_addr, const void* g) {
  asm volatile("global_load_async_to_lds_b128 %0, %1, off"
               :: "v"(lds_addr), "v"(g) : "memory");
  asm volatile("global_load_async_to_lds_b128 %0, %1, off offset:16"
               :: "v"(lds_addr), "v"(g) : "memory");
  asm volatile("global_load_async_to_lds_b128 %0, %1, off offset:32"
               :: "v"(lds_addr), "v"(g) : "memory");
  asm volatile("global_load_async_to_lds_b128 %0, %1, off offset:48"
               :: "v"(lds_addr), "v"(g) : "memory");
}

// ---------------------------------------------------------------------------
// f32 -> f16 convert (contiguous)
// ---------------------------------------------------------------------------
__global__ __launch_bounds__(256)
void cvt_f32_to_f16(const float* __restrict__ in, _Float16* __restrict__ out, int n)
{
  int i = (blockIdx.x * 256 + threadIdx.x) * 4;
  if (i < n) {
    float4 v = *(const float4*)(in + i);
    v4h h = { (_Float16)v.x, (_Float16)v.y, (_Float16)v.z, (_Float16)v.w };
    *(v4h*)(out + i) = h;
  }
}

// ---------------------------------------------------------------------------
// f32 [R][C] -> f16 [C][R] transpose via LDS tile (R, C multiples of 32)
// ---------------------------------------------------------------------------
__global__ __launch_bounds__(256)
void transpose_cvt(const float* __restrict__ in, _Float16* __restrict__ out,
                   int R, int C)
{
  __shared__ float tile[32][33];
  const int cb = blockIdx.x * 32, rb = blockIdx.y * 32;
  const int tx = threadIdx.x, ty = threadIdx.y;
#pragma unroll
  for (int i = ty; i < 32; i += 8)
    tile[i][tx] = in[(size_t)(rb + i) * C + cb + tx];
  __syncthreads();
#pragma unroll
  for (int i = ty; i < 32; i += 8)
    out[(size_t)(cb + i) * R + rb + tx] = (_Float16)tile[tx][i];
}

// ---------------------------------------------------------------------------
// WMMA GEMM:  C[M][N] = A[M][K] (f16, row major) * Bt[N][K]^T (f16) + bias
// mode 0: scatter f16 into q/k/v [B,H,L,128];  mode 1: f32 out[M][N]
// block: 256 thr = 8 waves (4x2), block tile 128x128, wave tile 32x64.
// K staged 64-deep with GLOBAL_LOAD_ASYNC_TO_LDS_B128, double buffered,
// branchless steady-state loop (last iteration peeled), s_wait_asynccnt sync.
// 16 WMMAs per barrier pair.
// ---------------------------------------------------------------------------
__global__ __launch_bounds__(256)
void wmma_gemm(const _Float16* __restrict__ A, const _Float16* __restrict__ Bt,
               const float* __restrict__ bias, int M, int N, int K, int mode,
               _Float16* __restrict__ qh, _Float16* __restrict__ kh,
               _Float16* __restrict__ vh, float* __restrict__ outp)
{
  __shared__ _Float16 lA[2][128 * 64];   // [buf][m][k]  16KB each
  __shared__ _Float16 lB[2][128 * 64];   // [buf][n][k]  16KB each
  const int tid  = threadIdx.x;
  const int lane = tid & 31, w = tid >> 5;
  const int wm = w >> 1, wn = w & 1;
  const int m0 = blockIdx.y * 128, n0 = blockIdx.x * 128;
  const int half = lane >> 4, l16 = lane & 15;

  const v8f zf = {};
  v8f c[2][4];
#pragma unroll
  for (int i = 0; i < 2; ++i)
#pragma unroll
    for (int j = 0; j < 4; ++j) c[i][j] = zf;

  // staging: each thread DMAs one 64B segment of A and one of B per stage
  const int srow = tid >> 1, skp = (tid & 1) * 32;
  const _Float16* agbase = A  + (size_t)(m0 + srow) * K + skp;
  const _Float16* bgbase = Bt + (size_t)(n0 + srow) * K + skp;
  const uint32_t laA0 = (uint32_t)(uintptr_t)&lA[0][srow * 64 + skp];
  const uint32_t laA1 = (uint32_t)(uintptr_t)&lA[1][srow * 64 + skp];
  const uint32_t laB0 = (uint32_t)(uintptr_t)&lB[0][srow * 64 + skp];
  const uint32_t laB1 = (uint32_t)(uintptr_t)&lB[1][srow * 64 + skp];

  const int nk = K / 64;

  auto compute = [&](int buf) {
    const _Float16* a_ = lA[buf];
    const _Float16* b_ = lB[buf];
#pragma unroll
    for (int ks = 0; ks < 2; ++ks) {
      v16h af[2], bf[4];
#pragma unroll
      for (int ms = 0; ms < 2; ++ms) {
        const int m = wm * 32 + ms * 16 + l16;
        const int o = m * 64 + ks * 32;
        af[ms] = hcat(*(const v8h*)&a_[o + half * 8],
                      *(const v8h*)&a_[o + 16 + half * 8]);
      }
#pragma unroll
      for (int ns = 0; ns < 4; ++ns) {
        const int n = wn * 64 + ns * 16 + l16;
        const int o = n * 64 + ks * 32;
        bf[ns] = hcat(*(const v8h*)&b_[o + half * 8],
                      *(const v8h*)&b_[o + 16 + half * 8]);
      }
#pragma unroll
      for (int ms = 0; ms < 2; ++ms)
#pragma unroll
        for (int ns = 0; ns < 4; ++ns)
          c[ms][ns] = WMMA_F16(af[ms], bf[ns], c[ms][ns]);
    }
  };

  // prologue: stage tile 0 into buffer 0
  async_copy64(laA0, agbase);
  async_copy64(laB0, bgbase);

  // steady state: branchless prefetch of stage j+1, 16 WMMAs per barrier pair
  for (int j = 0; j < nk - 1; ++j) {
    const int nbuf = (j + 1) & 1;
    const uint32_t la = nbuf ? laA1 : laA0;
    const uint32_t lb = nbuf ? laB1 : laB0;
    const int koff = (j + 1) * 64;
    async_copy64(la, agbase + koff);
    async_copy64(lb, bgbase + koff);
    asm volatile("s_wait_asynccnt 0x8" ::: "memory");
    __syncthreads();
    compute(j & 1);
    __syncthreads();
  }
  // epilogue iteration
  asm volatile("s_wait_asynccnt 0x0" ::: "memory");
  __syncthreads();
  compute((nk - 1) & 1);

#pragma unroll
  for (int ms = 0; ms < 2; ++ms) {
#pragma unroll
    for (int ns = 0; ns < 4; ++ns) {
      const int ng = n0 + wn * 64 + ns * 16 + l16;
      const float bv = bias[ng];
#pragma unroll
      for (int i = 0; i < 8; ++i) {
        const int mg = m0 + wm * 32 + ms * 16 + i + half * 8;
        const float v = c[ms][ns][i] + bv;
        if (mode == 0) {
          const int which = ng >> 11;        // 0=q, 1=k, 2=v
          const int hh = (ng >> 7) & (H_ - 1);
          const int dd = ng & (HD_ - 1);
          const int b  = mg >> 11;           // token / L
          const int l  = mg & (L_ - 1);
          _Float16* p = (which == 0) ? qh : (which == 1) ? kh : vh;
          p[(((size_t)(b * H_ + hh)) * L_ + l) * HD_ + dd] = (_Float16)v;
        } else {
          outp[(size_t)mg * N + ng] = v;
        }
      }
    }
  }
}

// ---------------------------------------------------------------------------
// RMSNorm + RoPE over q and k head rows.  1 wave per 128-dim row.
// ---------------------------------------------------------------------------
__global__ __launch_bounds__(128)
void norm_rope(_Float16* __restrict__ Qh, _Float16* __restrict__ Kh,
               const float* __restrict__ qsc, const float* __restrict__ ksc)
{
  const int row  = blockIdx.x * 4 + (threadIdx.x >> 5);
  const int lane = threadIdx.x & 31;
  const int total = B_ * H_ * L_;
  const int sel = (row >= total);
  const int r   = sel ? row - total : row;
  const int pos = r & (L_ - 1);
  _Float16* p = (sel ? Kh : Qh) + (size_t)r * HD_;
  const float* sc = sel ? ksc : qsc;

  const int d = lane * 2;
  float x0 = (float)p[d],      x1 = (float)p[d + 1];
  float y0 = (float)p[d + 64], y1 = (float)p[d + 65];
  float ss = x0 * x0 + x1 * x1 + y0 * y0 + y1 * y1;
#pragma unroll
  for (int off = 16; off > 0; off >>= 1) ss += __shfl_xor(ss, off, 32);
  const float rs = rsqrtf(ss * (1.0f / 128.0f) + 1e-6f);
  x0 *= rs * sc[d];      x1 *= rs * sc[d + 1];
  y0 *= rs * sc[d + 64]; y1 *= rs * sc[d + 65];

  const float lg = 9.210340371976184f;  // ln(10000)
  const float f0 = __expf(-((float)d       / 64.f) * lg);
  const float f1 = __expf(-((float)(d + 1) / 64.f) * lg);
  const float a0 = pos * f0, a1 = pos * f1;
  const float c0 = cosf(a0), s0 = sinf(a0);
  const float c1 = cosf(a1), s1 = sinf(a1);
  p[d]      = (_Float16)(x0 * c0 - y0 * s0);
  p[d + 1]  = (_Float16)(x1 * c1 - y1 * s1);
  p[d + 64] = (_Float16)(x0 * s0 + y0 * c0);
  p[d + 65] = (_Float16)(x1 * s1 + y1 * c1);
}

// ---------------------------------------------------------------------------
// Flash attention, causal.  grid (L/64, B*H), block 128 = 4 waves.
// Wave owns 16 q rows; KV blocks of 32; WMMA f16->f32.
// ---------------------------------------------------------------------------
__global__ __launch_bounds__(128)
void flash_attn(const _Float16* __restrict__ Qh, const _Float16* __restrict__ Kh,
                const _Float16* __restrict__ Vh, _Float16* __restrict__ Oh)
{
  __shared__ _Float16 Vt[HD_ * 32];     // V transposed [d][kv]   (8 KB)
  __shared__ _Float16 Pl[4][16 * 32];   // per-wave P tile        (4 KB)
  const int tid  = threadIdx.x;
  const int lane = tid & 31, w = tid >> 5;
  const int half = lane >> 4, l16 = lane & 15;
  const int bh = blockIdx.y;
  const int q0 = blockIdx.x * 64;
  const int qb = q0 + w * 16;
  const size_t base = (size_t)bh * L_ * HD_;
  const _Float16* Qp = Qh + base;
  const _Float16* Kp = Kh + base;
  const _Float16* Vp = Vh + base;

  v16h qa[4];
#pragma unroll
  for (int c2 = 0; c2 < 4; ++c2) {
    const _Float16* g = Qp + (size_t)(qb + l16) * HD_ + c2 * 32 + half * 8;
    qa[c2] = hcat(*(const v8h*)g, *(const v8h*)(g + 16));
  }

  const v8f zf = {};
  float mst[8], lst[8];
  v8f o[8];
#pragma unroll
  for (int i = 0; i < 8; ++i) { mst[i] = -1e30f; lst[i] = 0.f; o[i] = zf; }

  const int qlast = qb + 15;
  const int jmax  = (q0 + 63) >> 5;
  const int vr = tid >> 2, vd0 = (tid & 3) * 32;

  for (int j = 0; j <= jmax; ++j) {
    const int kv0 = j * 32;
    __syncthreads();
#pragma unroll
    for (int d8 = 0; d8 < 4; ++d8) {
      v8h vv = *(const v8h*)(Vp + (size_t)(kv0 + vr) * HD_ + vd0 + d8 * 8);
#pragma unroll
      for (int e = 0; e < 8; ++e) Vt[(vd0 + d8 * 8 + e) * 32 + vr] = vv[e];
    }
    __syncthreads();

    if (kv0 <= qlast) {                    // wave-uniform
      v8f s0 = {}, s1 = {};
#pragma unroll
      for (int c2 = 0; c2 < 4; ++c2) {
        const _Float16* g0 = Kp + (size_t)(kv0 + l16) * HD_      + c2 * 32 + half * 8;
        const _Float16* g1 = Kp + (size_t)(kv0 + 16 + l16) * HD_ + c2 * 32 + half * 8;
        v16h b0 = hcat(*(const v8h*)g0, *(const v8h*)(g0 + 16));
        v16h b1 = hcat(*(const v8h*)g1, *(const v8h*)(g1 + 16));
        s0 = WMMA_F16(qa[c2], b0, s0);
        s1 = WMMA_F16(qa[c2], b1, s1);
      }

      const float scl = 0.08838834764831845f;   // 1/sqrt(128)
      float fz[8];
#pragma unroll
      for (int i = 0; i < 8; ++i) {
        const int qg = qb + i + half * 8;
        const int kc0 = kv0 + l16, kc1 = kc0 + 16;
        const float a0 = (kc0 <= qg) ? s0[i] * scl : -1e30f;
        const float a1 = (kc1 <= qg) ? s1[i] * scl : -1e30f;
        float rm = fmaxf(a0, a1);
#pragma unroll
        for (int off = 1; off < 16; off <<= 1) rm = fmaxf(rm, __shfl_xor(rm, off, 32));
        const float mn = fmaxf(mst[i], rm);
        const float p0 = (kc0 <= qg) ? __expf(a0 - mn) : 0.f;
        const float p1 = (kc1 <= qg) ? __expf(a1 - mn) : 0.f;
        float rsum = p0 + p1;
#pragma unroll
        for (int off = 1; off < 16; off <<= 1) rsum += __shfl_xor(rsum, off, 32);
        const float f = __expf(mst[i] - mn);
        lst[i] = lst[i] * f + rsum;
        mst[i] = mn;
        fz[i]  = f;
        const int rr = i + half * 8;
        Pl[w][rr * 32 + l16]      = (_Float16)p0;
        Pl[w][rr * 32 + 16 + l16] = (_Float16)p1;
      }
#pragma unroll
      for (int t = 0; t < 8; ++t)
#pragma unroll
        for (int i = 0; i < 8; ++i) o[t][i] *= fz[i];

      v16h pa = hcat(*(const v8h*)&Pl[w][l16 * 32 + half * 8],
                     *(const v8h*)&Pl[w][l16 * 32 + 16 + half * 8]);
#pragma unroll
      for (int t = 0; t < 8; ++t) {
        const int d = t * 16 + l16;
        v16h vf = hcat(*(const v8h*)&Vt[d * 32 + half * 8],
                       *(const v8h*)&Vt[d * 32 + 16 + half * 8]);
        o[t] = WMMA_F16(pa, vf, o[t]);
      }
    }
  }

  const int b = bh >> 4, hh = bh & (H_ - 1);
#pragma unroll
  for (int i = 0; i < 8; ++i) {
    const int qg = qb + i + half * 8;
    const float inv = 1.0f / lst[i];
#pragma unroll
    for (int t = 0; t < 8; ++t) {
      const float v = o[t][i] * inv;
      Oh[((size_t)(b * L_ + qg)) * D_ + hh * HD_ + t * 16 + l16] = (_Float16)v;
    }
  }
}

// ---------------------------------------------------------------------------
extern "C" void kernel_launch(void* const* d_in, const int* in_sizes, int n_in,
                              void* d_out, int out_size, void* d_ws, size_t ws_size,
                              hipStream_t stream)
{
  (void)in_sizes; (void)n_in; (void)out_size; (void)ws_size;
  const float* x     = (const float*)d_in[0];
  const float* w_qkv = (const float*)d_in[1];
  const float* b_qkv = (const float*)d_in[2];
  const float* q_s   = (const float*)d_in[3];
  const float* k_s   = (const float*)d_in[4];
  const float* w_out = (const float*)d_in[5];
  const float* b_out = (const float*)d_in[6];
  float* outp = (float*)d_out;

  char* ws = (char*)d_ws;
  auto alloc = [&](size_t bytes) {
    char* p = ws;
    ws += (bytes + 255) & ~(size_t)255;
    return p;
  };
  _Float16* Xh    = (_Float16*)alloc((size_t)TOK * D_ * 2);
  _Float16* WqkvT = (_Float16*)alloc((size_t)N3  * D_ * 2);
  _Float16* WoutT = (_Float16*)alloc((size_t)D_  * D_ * 2);
  _Float16* Qh    = (_Float16*)alloc((size_t)TOK * D_ * 2);
  _Float16* Kh    = (_Float16*)alloc((size_t)TOK * D_ * 2);
  _Float16* Vh    = (_Float16*)alloc((size_t)TOK * D_ * 2);
  _Float16* Oh    = (_Float16*)alloc((size_t)TOK * D_ * 2);

  cvt_f32_to_f16<<<(TOK * D_ / 4 + 255) / 256, 256, 0, stream>>>(x, Xh, TOK * D_);
  transpose_cvt<<<dim3(N3 / 32, D_ / 32), dim3(32, 8), 0, stream>>>(w_qkv, WqkvT, D_, N3);
  transpose_cvt<<<dim3(D_ / 32, D_ / 32), dim3(32, 8), 0, stream>>>(w_out, WoutT, D_, D_);

  wmma_gemm<<<dim3(N3 / 128, TOK / 128), 256, 0, stream>>>(
      Xh, WqkvT, b_qkv, TOK, N3, D_, 0, Qh, Kh, Vh, nullptr);

  norm_rope<<<(2 * B_ * H_ * L_) / 4, 128, 0, stream>>>(Qh, Kh, q_s, k_s);

  flash_attn<<<dim3(L_ / 64, B_ * H_), 128, 0, stream>>>(Qh, Kh, Vh, Oh);

  wmma_gemm<<<dim3(D_ / 128, TOK / 128), 256, 0, stream>>>(
      Oh, WoutT, b_out, TOK, D_, D_, 1, nullptr, nullptr, nullptr, outp);
}
